// Decoder_81080392614298
// MI455X (gfx1250) — matrix operand order repaired and checked by reading, not compiled, not run on hardware
//
#include <hip/hip_runtime.h>
#include <math.h>

typedef __attribute__((ext_vector_type(2)))  float  v2f;
typedef __attribute__((ext_vector_type(8)))  float  v8f;
typedef __attribute__((ext_vector_type(16))) __bf16 v16bf;

#define Hd   1024
#define Bd   64
#define Ld   32
#define Vd   32000
#define G3H  3072
#define ROWS 2048            // L*B
#define KT   16              // K tile for fp32 path
#define LDT  18              // padded LDS stride (floats)

// ---------------------------------------------------------------------------
// Embedding gather + ReLU
// ---------------------------------------------------------------------------
__global__ void embed_relu_kernel(const int* __restrict__ target,
                                  const float* __restrict__ emb,
                                  float* __restrict__ x_all) {
    const int row = blockIdx.x;          // 0..2047
    const int t = row >> 6, b = row & 63;
    const int tok = (t == 0) ? 1 : target[(t - 1) * Bd + b];
    const float* src = emb + (size_t)tok * Hd;
    float* dst = x_all + (size_t)row * Hd;
    for (int c = threadIdx.x; c < Hd; c += blockDim.x) {
        float v = src[c];
        dst[c] = v > 0.f ? v : 0.f;
    }
}

// ---------------------------------------------------------------------------
// bf16x3 split-precision GEMM: C[M,N] = A[M,K] * W[N,K]^T + bias[N]
// fp32 inputs are split on the fly: x = hi(bf16) + lo(bf16), and
// C ~= Ah*Wh + Ah*Wl + Al*Wh  accumulated in fp32 via V_WMMA_F32_16X16X32_BF16.
// Tiles are scattered into LDS directly in WMMA fragment order so each lane
// reads its v16bf operand contiguously.
// Block tile 64x64, 8 waves; wave (wm=w&3, wn=w>>2) owns two 16x16 tiles.
// Requires 64|M, 64|N, 32|K.
// ---------------------------------------------------------------------------
__device__ __forceinline__ void scatterA(__bf16* hi, __bf16* lo, int row, int k, float4 v) {
    const int m = row & 15, mt = row >> 4;
    const int lane = (((k >> 3) & 1) << 4) + m;
    const int j0 = (k & 7) + ((k & 16) ? 8 : 0);
    const int base = mt * 512 + lane * 16 + j0;
    float f[4] = {v.x, v.y, v.z, v.w};
#pragma unroll
    for (int i = 0; i < 4; ++i) {
        __bf16 h = (__bf16)f[i];
        hi[base + i] = h;
        lo[base + i] = (__bf16)(f[i] - (float)h);
    }
}

__device__ __forceinline__ void scatterB(__bf16* hi, __bf16* lo, int row, int k, float4 v) {
    const int n = row & 15, nt = row >> 4;
    const int lane = (((k >> 3) & 1) << 4) + n;
    const int j0 = (k & 7) + ((k & 16) ? 8 : 0);
    const int base = nt * 512 + lane * 16 + j0;
    float f[4] = {v.x, v.y, v.z, v.w};
#pragma unroll
    for (int i = 0; i < 4; ++i) {
        __bf16 h = (__bf16)f[i];
        hi[base + i] = h;
        lo[base + i] = (__bf16)(f[i] - (float)h);
    }
}

__global__ __launch_bounds__(256)
void gemm_wmma_bf16x3(const float* __restrict__ A,
                      const float* __restrict__ W,
                      const float* __restrict__ bias,
                      float* __restrict__ C,
                      int N, int K) {
    __shared__ __align__(32) __bf16 Ah[64 * 32], Al[64 * 32];
    __shared__ __align__(32) __bf16 Wh[64 * 32], Wl[64 * 32];

    const int tid  = threadIdx.x;
    const int lane = tid & 31;
    const int wave = tid >> 5;
    const int wm   = wave & 3;       // M tile 0..3
    const int wn   = wave >> 2;      // N group 0..1 (two 16-col tiles)
    const int m0   = blockIdx.y * 64;
    const int n0   = blockIdx.x * 64;

    const int lrow = tid >> 2;           // 0..63 (row within tile)
    const int kq   = (tid & 3) * 4;      // 0,4,8,12

    v8f acc0 = {}; v8f acc1 = {};

    const float* gA = A + (size_t)(m0 + lrow) * K + kq;
    const float* gW = W + (size_t)(n0 + lrow) * K + kq;

    const __bf16* fAh = Ah + wm * 512 + lane * 16;
    const __bf16* fAl = Al + wm * 512 + lane * 16;
    const __bf16* fB0h = Wh + (wn * 2) * 512 + lane * 16;
    const __bf16* fB0l = Wl + (wn * 2) * 512 + lane * 16;
    const __bf16* fB1h = Wh + (wn * 2 + 1) * 512 + lane * 16;
    const __bf16* fB1l = Wl + (wn * 2 + 1) * 512 + lane * 16;

    for (int k0 = 0; k0 < K; k0 += 32) {
        __syncthreads();
        float4 a1 = *(const float4*)(gA + k0);
        float4 a2 = *(const float4*)(gA + k0 + 16);
        float4 w1 = *(const float4*)(gW + k0);
        float4 w2 = *(const float4*)(gW + k0 + 16);
        if (k0 + 32 < K) {
            __builtin_prefetch(gA + k0 + 32, 0, 0);
            __builtin_prefetch(gW + k0 + 32, 0, 0);
        }
        scatterA(Ah, Al, lrow, kq,      a1);
        scatterA(Ah, Al, lrow, kq + 16, a2);
        scatterB(Wh, Wl, lrow, kq,      w1);
        scatterB(Wh, Wl, lrow, kq + 16, w2);
        __syncthreads();

        v16bf ah  = *(const v16bf*)fAh;
        v16bf al  = *(const v16bf*)fAl;
        v16bf b0h = *(const v16bf*)fB0h;
        v16bf b0l = *(const v16bf*)fB0l;
        v16bf b1h = *(const v16bf*)fB1h;
        v16bf b1l = *(const v16bf*)fB1l;

        acc0 = __builtin_amdgcn_wmma_f32_16x16x32_bf16(false, ah, false, b0h, (short)0, acc0, false, false);
        acc0 = __builtin_amdgcn_wmma_f32_16x16x32_bf16(false, ah, false, b0l, (short)0, acc0, false, false);
        acc0 = __builtin_amdgcn_wmma_f32_16x16x32_bf16(false, al, false, b0h, (short)0, acc0, false, false);
        acc1 = __builtin_amdgcn_wmma_f32_16x16x32_bf16(false, ah, false, b1h, (short)0, acc1, false, false);
        acc1 = __builtin_amdgcn_wmma_f32_16x16x32_bf16(false, ah, false, b1l, (short)0, acc1, false, false);
        acc1 = __builtin_amdgcn_wmma_f32_16x16x32_bf16(false, al, false, b1h, (short)0, acc1, false, false);
    }

    const int hi = lane >> 4, lr = lane & 15;
    const int col0 = n0 + wn * 32 + lr;
    const int col1 = col0 + 16;
    const float bia0 = bias[col0], bia1 = bias[col1];
#pragma unroll
    for (int i = 0; i < 8; ++i) {
        const int row = m0 + wm * 16 + hi * 8 + i;
        C[(size_t)row * N + col0] = acc0[i] + bia0;
        C[(size_t)row * N + col1] = acc1[i] + bia1;
    }
}

// ---------------------------------------------------------------------------
// One GRU step: gh = h_prev @ w_hh^T in exact fp32 WMMA (recurrence error must
// not compound), fused gates with precomputed gi; h_new -> outs[t].
// Grid: H/32 = 32 blocks x 256 threads. Each wave owns one 16x16 tile per gate.
// ---------------------------------------------------------------------------
__global__ __launch_bounds__(256)
void gru_step_wmma(const float* __restrict__ hprev,   // [64,1024]
                   const float* __restrict__ w_hh,    // [3072,1024]
                   const float* __restrict__ b_hh,    // [3072]
                   const float* __restrict__ gi,      // [64,3072], includes b_ih
                   float* __restrict__ hnew) {        // [64,1024]
    __shared__ __align__(16) float At[64 * LDT];
    __shared__ __align__(16) float Wg[3][32 * LDT];

    const int tid  = threadIdx.x;
    const int lane = tid & 31;
    const int wave = tid >> 5;
    const int wm   = wave & 3;           // M tile over B=64
    const int wn   = wave >> 2;          // 0/1 : 16-col group within the 32-col slice
    const int n0   = blockIdx.x * 32;    // column slice of H

    const int arow = tid >> 2;           // 0..63, A loads
    const int acol = (tid & 3) * 4;
    const int wrow = tid >> 3;           // 0..31, W loads
    const int wcol = (tid & 7) * 2;
    const int hi   = lane >> 4;
    const int lr   = lane & 15;

    v8f acc[3] = {};

    const float* gA = hprev + (size_t)arow * Hd + acol;

    for (int k0 = 0; k0 < Hd; k0 += KT) {
        __syncthreads();
        float4 va = *(const float4*)(gA + k0);
        float* sa = At + arow * LDT + acol;
        sa[0] = va.x; sa[1] = va.y; sa[2] = va.z; sa[3] = va.w;
#pragma unroll
        for (int g = 0; g < 3; ++g) {
            const float* gw = w_hh + (size_t)(g * Hd + n0 + wrow) * Hd + k0 + wcol;
            float2 vw = *(const float2*)gw;
            float* sw = Wg[g] + wrow * LDT + wcol;
            sw[0] = vw.x; sw[1] = vw.y;
        }
        __syncthreads();

#pragma unroll
        for (int kk = 0; kk < KT; kk += 4) {
            v2f a = *(const v2f*)(At + (wm * 16 + lr) * LDT + kk + 2 * hi);
#pragma unroll
            for (int g = 0; g < 3; ++g) {
                v2f b = *(const v2f*)(Wg[g] + (wn * 16 + lr) * LDT + kk + 2 * hi);
                acc[g] = __builtin_amdgcn_wmma_f32_16x16x4_f32(false, a, false, b,
                                                               (short)0, acc[g], false, false);
            }
        }
    }

    const int c = n0 + wn * 16 + lr;     // column within H
    const float bhr = b_hh[c], bhz = b_hh[Hd + c], bhn = b_hh[2 * Hd + c];
#pragma unroll
    for (int i = 0; i < 8; ++i) {
        const int b = wm * 16 + hi * 8 + i;   // batch row
        const float gir = gi[(size_t)b * G3H + c];
        const float giz = gi[(size_t)b * G3H + Hd + c];
        const float gin = gi[(size_t)b * G3H + 2 * Hd + c];
        const float r = 1.f / (1.f + expf(-(gir + acc[0][i] + bhr)));
        const float z = 1.f / (1.f + expf(-(giz + acc[1][i] + bhz)));
        const float n = tanhf(gin + r * (acc[2][i] + bhn));
        const float hp = hprev[(size_t)b * Hd + c];
        hnew[(size_t)b * Hd + c] = (1.f - z) * n + z * hp;
    }
}

// ---------------------------------------------------------------------------
// Per-row online log-sum-exp over V -> stats[row] = {max, log(sum)}
// ---------------------------------------------------------------------------
__global__ __launch_bounds__(256)
void row_logsumexp(const float* __restrict__ logits, float* __restrict__ stats) {
    __shared__ float sm[256], ss[256];
    const int row = blockIdx.x;
    const float* p = logits + (size_t)row * Vd;
    float m = -3.4e38f, s = 0.f;
    for (int v = threadIdx.x; v < Vd; v += 256) {
        const float x = p[v];
        if (x > m) { s = s * expf(m - x) + 1.f; m = x; }
        else       { s += expf(x - m); }
    }
    sm[threadIdx.x] = m; ss[threadIdx.x] = s;
    __syncthreads();
    for (int off = 128; off > 0; off >>= 1) {
        if (threadIdx.x < off) {
            const float m1 = sm[threadIdx.x], s1 = ss[threadIdx.x];
            const float m2 = sm[threadIdx.x + off], s2 = ss[threadIdx.x + off];
            const float M = m1 > m2 ? m1 : m2;
            ss[threadIdx.x] = s1 * expf(m1 - M) + s2 * expf(m2 - M);
            sm[threadIdx.x] = M;
        }
        __syncthreads();
    }
    if (threadIdx.x == 0) {
        stats[row * 2 + 0] = sm[0];
        stats[row * 2 + 1] = logf(ss[0]);
    }
}

__global__ __launch_bounds__(256)
void softmax_apply(float* __restrict__ logits, const float* __restrict__ stats) {
    const size_t idx4 = ((size_t)blockIdx.x * 256 + threadIdx.x) * 4;
    const int row = (int)(idx4 / Vd);
    const float sub = stats[row * 2] + stats[row * 2 + 1];
    float4 v = *(float4*)(logits + idx4);
    v.x -= sub; v.y -= sub; v.z -= sub; v.w -= sub;
    *(float4*)(logits + idx4) = v;
}

// ---------------------------------------------------------------------------
extern "C" void kernel_launch(void* const* d_in, const int* in_sizes, int n_in,
                              void* d_out, int out_size, void* d_ws, size_t ws_size,
                              hipStream_t stream) {
    (void)in_sizes; (void)n_in; (void)out_size; (void)ws_size;
    const int*   target = (const int*)  d_in[0];
    // d_in[1] encoder_outputs: unused by the reference
    const float* hidden = (const float*)d_in[2];
    const float* emb_w  = (const float*)d_in[3];
    const float* w_ih   = (const float*)d_in[4];
    const float* w_hh   = (const float*)d_in[5];
    const float* b_ih   = (const float*)d_in[6];
    const float* b_hh   = (const float*)d_in[7];
    const float* fc_w   = (const float*)d_in[8];
    const float* fc_b   = (const float*)d_in[9];
    float* out = (float*)d_out;

    // workspace layout (floats)
    float* x_all  = (float*)d_ws;                      // 2048*1024
    float* gi_all = x_all  + (size_t)ROWS * Hd;        // 2048*3072
    float* outs   = gi_all + (size_t)ROWS * G3H;       // 2048*1024
    float* stats  = outs   + (size_t)ROWS * Hd;        // 2048*2

    // 1) embedding gather + relu for all steps
    embed_relu_kernel<<<ROWS, 256, 0, stream>>>(target, emb_w, x_all);

    // 2) batched input-side GEMM (bf16x3): gi_all = x_all @ w_ih^T + b_ih
    gemm_wmma_bf16x3<<<dim3(G3H / 64, ROWS / 64), 256, 0, stream>>>(
        x_all, w_ih, b_ih, gi_all, G3H, Hd);

    // 3) sequential GRU steps (h_t lives in outs[t]); exact fp32 WMMA
    for (int t = 0; t < Ld; ++t) {
        const float* hprev = (t == 0) ? hidden : (outs + (size_t)(t - 1) * Bd * Hd);
        gru_step_wmma<<<Hd / 32, 256, 0, stream>>>(
            hprev, w_hh, b_hh, gi_all + (size_t)t * Bd * G3H,
            outs + (size_t)t * Bd * Hd);
    }

    // 4) output projection (bf16x3): logits = outs @ fc_w^T + fc_b -> d_out
    gemm_wmma_bf16x3<<<dim3(Vd / 64, ROWS / 64), 256, 0, stream>>>(
        outs, fc_w, fc_b, out, Vd, Hd);

    // 5) log_softmax in place
    row_logsumexp<<<ROWS, 256, 0, stream>>>(out, stats);
    softmax_apply<<<(ROWS * (size_t)Vd) / (256 * 4), 256, 0, stream>>>(out, stats);

    // 6) h_last = outs[L-1] appended after log_probs
    hipMemcpyAsync(out + (size_t)ROWS * Vd,
                   outs + (size_t)(Ld - 1) * Bd * Hd,
                   (size_t)Bd * Hd * sizeof(float),
                   hipMemcpyDeviceToDevice, stream);
}
